// EncoderBlock_67688684585159
// MI455X (gfx1250) — matrix operand order repaired and checked by reading, not compiled, not run on hardware
//
#include <hip/hip_runtime.h>
#include <stdint.h>

typedef __attribute__((ext_vector_type(16))) __bf16 v16bf;
typedef __attribute__((ext_vector_type(8)))  float  v8f;

constexpr int BATCH  = 2;
constexpr int SEQ    = 2048;
constexpr int DMODEL = 1024;
constexpr int NHEAD  = 16;
constexpr int DKH    = 64;
constexpr int DFF    = 4096;
constexpr int TOK    = BATCH * SEQ;   // 4096

union Frag {
  uint4 q[2];
  v16bf v;
  unsigned short u[16];
};

__device__ __forceinline__ unsigned short f2bf(float f) {
  unsigned int u = __float_as_uint(f);
  u += 0x7FFFu + ((u >> 16) & 1u);          // round-to-nearest-even
  return (unsigned short)(u >> 16);
}

__device__ __forceinline__ v8f vzero8() {
  v8f z = {0.f, 0.f, 0.f, 0.f, 0.f, 0.f, 0.f, 0.f};
  return z;
}

// Low 32 bits of a generic pointer into the LDS aperture == LDS byte offset
// (ISA 10.2).
__device__ __forceinline__ unsigned lds_off(const void* p) {
  return (unsigned)(size_t)p;
}

// Async global -> LDS 16B copy, tracked with ASYNCcnt (VGLOBAL op 98).
__device__ __forceinline__ void async_ld_b128(unsigned lds_addr, const void* g) {
  asm volatile("global_load_async_to_lds_b128 %0, %1, off"
               :: "v"(lds_addr), "v"((unsigned long long)(size_t)g)
               : "memory");
}

__device__ __forceinline__ void wait_async0() {
  asm volatile("s_wait_asynccnt 0x0" ::: "memory");
}

// ---------------------------------------------------------------------------
// fp32 [K,N] -> bf16 transposed [N,K] (weights, once per launch)
// ---------------------------------------------------------------------------
__global__ __launch_bounds__(256) void cvt_bf16_t_kernel(
    const float* __restrict__ in, unsigned short* __restrict__ out, int K, int N) {
  int i = blockIdx.x * 256 + threadIdx.x;
  if (i < K * N) {
    int k = i / N, n = i % N;
    out[(size_t)n * K + k] = f2bf(in[i]);
  }
}

// ---------------------------------------------------------------------------
// LayerNorm (torch semantics: unbiased std, eps added to std) -> bf16 row
// ---------------------------------------------------------------------------
__global__ __launch_bounds__(256) void ln_bf16_kernel(
    const float* __restrict__ x, const float* __restrict__ alpha,
    const float* __restrict__ beta, unsigned short* __restrict__ out) {
  const int row = blockIdx.x;
  const float* xr = x + (size_t)row * DMODEL;
  float s = 0.f, s2 = 0.f;
  for (int i = threadIdx.x; i < DMODEL; i += 256) {
    float v = xr[i];
    s += v; s2 += v * v;
  }
#pragma unroll
  for (int off = 16; off > 0; off >>= 1) {
    s  += __shfl_down(s,  off, 32);
    s2 += __shfl_down(s2, off, 32);
  }
  __shared__ float red0[8], red1[8];
  if ((threadIdx.x & 31) == 0) { red0[threadIdx.x >> 5] = s; red1[threadIdx.x >> 5] = s2; }
  __syncthreads();
  float ts = 0.f, ts2 = 0.f;
#pragma unroll
  for (int w = 0; w < 8; ++w) { ts += red0[w]; ts2 += red1[w]; }
  const float mean = ts * (1.0f / DMODEL);
  const float var  = (ts2 - mean * ts) * (1.0f / (DMODEL - 1));  // unbiased
  const float inv  = alpha[0] / (sqrtf(fmaxf(var, 0.f)) + 1e-6f);
  const float bb   = beta[0];
  unsigned short* orow = out + (size_t)row * DMODEL;
  for (int i = threadIdx.x; i < DMODEL; i += 256)
    orow[i] = f2bf((xr[i] - mean) * inv + bb);
}

// ---------------------------------------------------------------------------
// WMMA GEMM, async double-buffered LDS pipeline.
//   C[M,N] = A[M,K] (bf16 row-major) @ Bt[N,K]^T (bf16, pre-transposed) + bias
// block = 256 threads (8 waves), tile 64x64, BK = 32; wave -> 16x32 of C.
// Per iter: s_wait_asynccnt 0 -> barrier -> issue async tile i+1 -> WMMA tile i.
// ---------------------------------------------------------------------------
enum GemmEpi { EPI_HEADS = 0, EPI_VT = 1, EPI_RELU = 2, EPI_RESID = 3 };

template <int EPI>
__global__ __launch_bounds__(256) void gemm_wmma_kernel(
    const unsigned short* __restrict__ A, const unsigned short* __restrict__ Bt,
    const float* __restrict__ bias, const float* __restrict__ resid,
    unsigned short* __restrict__ outb, float* __restrict__ outf,
    int M, int N, int K) {
  __shared__ __align__(16) unsigned short As[2][64][40];  // [buf][m][k], 80B rows (16B aligned)
  __shared__ __align__(16) unsigned short Bs[2][64][40];  // [buf][n][k]
  const int tid  = threadIdx.x;
  const int lane = tid & 31;
  const int wave = tid >> 5;
  const int m0   = blockIdx.y * 64;
  const int n0   = blockIdx.x * 64;
  const int mi   = wave & 3;     // M sub-tile
  const int ni   = wave >> 2;    // N strip (32 cols)
  const int half = lane >> 4;
  const int l16  = lane & 15;

  v8f c0 = vzero8(), c1 = vzero8();

  // staging map: one 16B async copy per thread per tile
  const int row = tid >> 2;          // 0..63
  const int cg  = (tid & 3) * 8;     // 0,8,16,24 (elements)
  const unsigned short* ga = A  + (size_t)(m0 + row) * K + cg;
  const unsigned short* gb = Bt + (size_t)(n0 + row) * K + cg;
  const unsigned lda0 = lds_off(&As[0][row][cg]);
  const unsigned lda1 = lds_off(&As[1][row][cg]);
  const unsigned ldb0 = lds_off(&Bs[0][row][cg]);
  const unsigned ldb1 = lds_off(&Bs[1][row][cg]);

  async_ld_b128(lda0, ga);
  async_ld_b128(ldb0, gb);

  const int am = mi * 16 + l16;
  const int bn = ni * 32 + l16;
  const int nsteps = K >> 5;

  for (int i = 0; i < nsteps; ++i) {
    wait_async0();          // this wave's loads for buffer i&1 have landed
    __syncthreads();        // everyone's loads landed; buffer (i+1)&1 is free
    if (i + 1 < nsteps) {
      const int k = (i + 1) << 5;
      async_ld_b128(((i + 1) & 1) ? lda1 : lda0, ga + k);
      async_ld_b128(((i + 1) & 1) ? ldb1 : ldb0, gb + k);
    }
    const int cur = i & 1;
    // A-frag: K = (e/8)*16 + half*8 + (e%8) -> two contiguous 16B chunks
    Frag af, b0, b1;
    af.q[0] = *(const uint4*)&As[cur][am][half * 8];
    af.q[1] = *(const uint4*)&As[cur][am][16 + half * 8];
    // B-frag: K = (e/8)*16 + half*8 + (e%8), N = lane%16 -> two 16B chunks of Bs rows
    b0.q[0] = *(const uint4*)&Bs[cur][bn][half * 8];
    b0.q[1] = *(const uint4*)&Bs[cur][bn][16 + half * 8];
    b1.q[0] = *(const uint4*)&Bs[cur][bn + 16][half * 8];
    b1.q[1] = *(const uint4*)&Bs[cur][bn + 16][16 + half * 8];

    c0 = __builtin_amdgcn_wmma_f32_16x16x32_bf16(false, af.v, false, b0.v, (short)0, c0, false, false);
    c1 = __builtin_amdgcn_wmma_f32_16x16x32_bf16(false, af.v, false, b1.v, (short)0, c1, false, false);
  }

  // C layout: VGPR r -> M = r + 8*half, N = lane%16
  const int gn0 = n0 + ni * 32 + l16;
  const int gn1 = gn0 + 16;
  const float bias0 = bias[gn0];
  const float bias1 = bias[gn1];
#pragma unroll
  for (int r = 0; r < 8; ++r) {
    const int gm = m0 + mi * 16 + 8 * half + r;
    float v0 = c0[r] + bias0;
    float v1 = c1[r] + bias1;
    if constexpr (EPI == EPI_HEADS || EPI == EPI_VT) {
      const int b = gm >> 11, s = gm & (SEQ - 1);     // M == TOK, SEQ == 2048
      const int h0 = gn0 >> 6, d0 = gn0 & 63;
      const int h1 = gn1 >> 6, d1 = gn1 & 63;
      if constexpr (EPI == EPI_HEADS) {               // [B,H,S,Dk]
        outb[(((size_t)(b * NHEAD + h0)) * SEQ + s) * DKH + d0] = f2bf(v0);
        outb[(((size_t)(b * NHEAD + h1)) * SEQ + s) * DKH + d1] = f2bf(v1);
      } else {                                        // V transposed: [B,H,Dk,S]
        outb[(((size_t)(b * NHEAD + h0)) * DKH + d0) * SEQ + s] = f2bf(v0);
        outb[(((size_t)(b * NHEAD + h1)) * DKH + d1) * SEQ + s] = f2bf(v1);
      }
    } else if constexpr (EPI == EPI_RELU) {
      outb[(size_t)gm * N + gn0] = f2bf(fmaxf(v0, 0.f));
      outb[(size_t)gm * N + gn1] = f2bf(fmaxf(v1, 0.f));
    } else {  // EPI_RESID: fp32 out = residual + C
      outf[(size_t)gm * N + gn0] = resid[(size_t)gm * N + gn0] + v0;
      outf[(size_t)gm * N + gn1] = resid[(size_t)gm * N + gn1] + v1;
    }
  }
}

// ---------------------------------------------------------------------------
// Flash-style attention: one wave per (b, h, 16-query tile); online softmax.
// Q,K in [B,H,S,Dk] bf16, V in [B,H,Dk,S] bf16. ctx out row-major [TOK,DMODEL].
// ---------------------------------------------------------------------------
__global__ __launch_bounds__(32) void attn_wmma_kernel(
    const unsigned short* __restrict__ Q, const unsigned short* __restrict__ Km,
    const unsigned short* __restrict__ Vt, const int* __restrict__ mask,
    unsigned short* __restrict__ ctx) {
  const int qt   = blockIdx.x & (SEQ / 16 - 1);   // 0..127
  const int bh   = blockIdx.x >> 7;               // 0..31
  const int b    = bh >> 4;
  const int h    = bh & 15;
  const int lane = threadIdx.x;
  const int half = lane >> 4;
  const int l16  = lane & 15;
  const int qbase = qt * 16;

  const unsigned short* Qb = Q  + ((size_t)bh * SEQ + qbase) * DKH;
  const unsigned short* Kb = Km + (size_t)bh * SEQ * DKH;
  const unsigned short* Vb = Vt + (size_t)bh * DKH * SEQ;

  __shared__ __align__(16) unsigned short p_lds[16][40];

  // Q A-fragments for K-dim chunks [0,32) and [32,64)
  Frag qf[2];
#pragma unroll
  for (int dc = 0; dc < 2; ++dc) {
    const unsigned short* src = Qb + (size_t)l16 * DKH + dc * 32 + half * 8;
    qf[dc].q[0] = *(const uint4*)(src);
    qf[dc].q[1] = *(const uint4*)(src + 16);
  }

  v8f ctxa[4];
#pragma unroll
  for (int j = 0; j < 4; ++j) ctxa[j] = vzero8();
  float rmax[8], rsum[8];
#pragma unroll
  for (int r = 0; r < 8; ++r) { rmax[r] = -3.0e38f; rsum[r] = 0.f; }

  for (int kb = 0; kb < SEQ; kb += 32) {
    // ---- scores S = Q @ K^T for 32 keys (two 16-key tiles) ----
    v8f sa[2];
    sa[0] = vzero8(); sa[1] = vzero8();
#pragma unroll
    for (int kt = 0; kt < 2; ++kt) {
      const int key = kb + kt * 16 + l16;           // this lane's N (key col)
      const unsigned short* kr = Kb + (size_t)key * DKH;
      Frag bf0, bf1;                                // K^T B-frags, dk = chunk + (e/8)*16 + half*8 + e%8
      bf0.q[0] = *(const uint4*)(kr + half * 8);
      bf0.q[1] = *(const uint4*)(kr + 16 + half * 8);
      bf1.q[0] = *(const uint4*)(kr + 32 + half * 8);
      bf1.q[1] = *(const uint4*)(kr + 32 + 16 + half * 8);
      sa[kt] = __builtin_amdgcn_wmma_f32_16x16x32_bf16(false, qf[0].v, false, bf0.v, (short)0, sa[kt], false, false);
      sa[kt] = __builtin_amdgcn_wmma_f32_16x16x32_bf16(false, qf[1].v, false, bf1.v, (short)0, sa[kt], false, false);
    }

    // ---- scale, mask, online softmax ----
    float pv[2][8];
    float chmax[8];
#pragma unroll
    for (int r = 0; r < 8; ++r) chmax[r] = -3.0e38f;
#pragma unroll
    for (int kt = 0; kt < 2; ++kt) {
#pragma unroll
      for (int r = 0; r < 8; ++r) {
        const int Mr  = r + 8 * half;
        const int col = kb + kt * 16 + l16;
        float sc = sa[kt][r] * 0.125f;              // 1/sqrt(64)
        const int mv = mask[(size_t)(qbase + Mr) * SEQ + col];
        sc = (mv == 0) ? -1e9f : sc;
        pv[kt][r] = sc;
        chmax[r] = fmaxf(chmax[r], sc);
      }
    }
#pragma unroll
    for (int r = 0; r < 8; ++r) {                   // row-max across 16 lanes of half
      float m = chmax[r];
      for (int off = 1; off < 16; off <<= 1) m = fmaxf(m, __shfl_xor(m, off, 32));
      const float nm    = fmaxf(rmax[r], m);
      const float scale = __expf(rmax[r] - nm);
      rmax[r] = nm;
      float lsum = 0.f;
#pragma unroll
      for (int kt = 0; kt < 2; ++kt) {
        float p = __expf(pv[kt][r] - nm);
        pv[kt][r] = p;
        lsum += p;
      }
      for (int off = 1; off < 16; off <<= 1) lsum += __shfl_xor(lsum, off, 32);
      rsum[r] = rsum[r] * scale + lsum;
#pragma unroll
      for (int j = 0; j < 4; ++j) ctxa[j][r] *= scale;
    }

    // ---- re-layout P (C-format f32) -> A-format bf16 via LDS ----
#pragma unroll
    for (int kt = 0; kt < 2; ++kt)
#pragma unroll
      for (int r = 0; r < 8; ++r)
        p_lds[r + 8 * half][kt * 16 + l16] = f2bf(pv[kt][r]);
    __syncthreads();
    Frag pf;
    const unsigned short* pr = &p_lds[l16][0];
    pf.q[0] = *(const uint4*)(pr + half * 8);
    pf.q[1] = *(const uint4*)(pr + 16 + half * 8);

    // ---- ctx += P @ V (V stored transposed: contiguous keys per dk row) ----
#pragma unroll
    for (int j = 0; j < 4; ++j) {
      const int dk = j * 16 + l16;
      const unsigned short* vr = Vb + (size_t)dk * SEQ + kb + half * 8;
      Frag vf;
      vf.q[0] = *(const uint4*)(vr);
      vf.q[1] = *(const uint4*)(vr + 16);
      ctxa[j] = __builtin_amdgcn_wmma_f32_16x16x32_bf16(false, pf.v, false, vf.v, (short)0, ctxa[j], false, false);
    }
    __syncthreads();
  }

  // ---- finalize: divide by softmax sum, store ctx row-major [TOK, DMODEL] ----
#pragma unroll
  for (int j = 0; j < 4; ++j) {
#pragma unroll
    for (int r = 0; r < 8; ++r) {
      const int Mr    = r + 8 * half;
      const int token = b * SEQ + qbase + Mr;
      const int col   = h * DKH + j * 16 + l16;
      ctx[(size_t)token * DMODEL + col] = f2bf(ctxa[j][r] / rsum[r]);
    }
  }
}

// ---------------------------------------------------------------------------
// Launcher
// ---------------------------------------------------------------------------
extern "C" void kernel_launch(void* const* d_in, const int* in_sizes, int n_in,
                              void* d_out, int out_size, void* d_ws, size_t ws_size,
                              hipStream_t stream) {
  (void)in_sizes; (void)n_in; (void)out_size; (void)ws_size;
  const float* x      = (const float*)d_in[0];
  const int*   mask   = (const int*)d_in[1];
  const float* Wq = (const float*)d_in[2];  const float* bq = (const float*)d_in[3];
  const float* Wk = (const float*)d_in[4];  const float* bk = (const float*)d_in[5];
  const float* Wv = (const float*)d_in[6];  const float* bv = (const float*)d_in[7];
  const float* Wo = (const float*)d_in[8];  const float* bo = (const float*)d_in[9];
  const float* W1 = (const float*)d_in[10]; const float* b1 = (const float*)d_in[11];
  const float* W2 = (const float*)d_in[12]; const float* b2 = (const float*)d_in[13];
  const float* alpha1 = (const float*)d_in[14]; const float* beta1 = (const float*)d_in[15];
  const float* alpha2 = (const float*)d_in[16]; const float* beta2 = (const float*)d_in[17];
  float* out = (float*)d_out;

  char* ws = (char*)d_ws;
  size_t off = 0;
  auto alloc = [&](size_t bytes) -> void* {
    void* p = ws + off;
    off += (bytes + 255) & ~(size_t)255;
    return p;
  };
  unsigned short* xnb  = (unsigned short*)alloc((size_t)TOK * DMODEL * 2);  // also LN2 out
  unsigned short* Wqt  = (unsigned short*)alloc((size_t)DMODEL * DMODEL * 2);  // [N][K]
  unsigned short* Wkt  = (unsigned short*)alloc((size_t)DMODEL * DMODEL * 2);
  unsigned short* Wvt  = (unsigned short*)alloc((size_t)DMODEL * DMODEL * 2);
  unsigned short* Wot  = (unsigned short*)alloc((size_t)DMODEL * DMODEL * 2);
  unsigned short* W1t  = (unsigned short*)alloc((size_t)DMODEL * DFF * 2);     // [DFF][DMODEL]
  unsigned short* W2t  = (unsigned short*)alloc((size_t)DFF * DMODEL * 2);     // [DMODEL][DFF]
  unsigned short* Qb   = (unsigned short*)alloc((size_t)TOK * DMODEL * 2);
  unsigned short* Kb   = (unsigned short*)alloc((size_t)TOK * DMODEL * 2);
  unsigned short* Vtb  = (unsigned short*)alloc((size_t)TOK * DMODEL * 2);
  unsigned short* ctxb = (unsigned short*)alloc((size_t)TOK * DMODEL * 2);
  float*          y1   = (float*)alloc((size_t)TOK * DMODEL * 4);
  unsigned short* hb   = Qb;  // FFN hidden (32MB) aliases dead Q/K/Vt/ctx region

  const int nW = DMODEL * DMODEL, nF = DMODEL * DFF;
  cvt_bf16_t_kernel<<<(nW + 255) / 256, 256, 0, stream>>>(Wq, Wqt, DMODEL, DMODEL);
  cvt_bf16_t_kernel<<<(nW + 255) / 256, 256, 0, stream>>>(Wk, Wkt, DMODEL, DMODEL);
  cvt_bf16_t_kernel<<<(nW + 255) / 256, 256, 0, stream>>>(Wv, Wvt, DMODEL, DMODEL);
  cvt_bf16_t_kernel<<<(nW + 255) / 256, 256, 0, stream>>>(Wo, Wot, DMODEL, DMODEL);
  cvt_bf16_t_kernel<<<(nF + 255) / 256, 256, 0, stream>>>(W1, W1t, DMODEL, DFF);
  cvt_bf16_t_kernel<<<(nF + 255) / 256, 256, 0, stream>>>(W2, W2t, DFF, DMODEL);

  // LN1
  ln_bf16_kernel<<<TOK, 256, 0, stream>>>(x, alpha1, beta1, xnb);

  // Q, K : [B,H,S,Dk]   V : [B,H,Dk,S]
  dim3 gProj(DMODEL / 64, TOK / 64);
  gemm_wmma_kernel<EPI_HEADS><<<gProj, 256, 0, stream>>>(xnb, Wqt, bq, nullptr, Qb,  nullptr, TOK, DMODEL, DMODEL);
  gemm_wmma_kernel<EPI_HEADS><<<gProj, 256, 0, stream>>>(xnb, Wkt, bk, nullptr, Kb,  nullptr, TOK, DMODEL, DMODEL);
  gemm_wmma_kernel<EPI_VT>   <<<gProj, 256, 0, stream>>>(xnb, Wvt, bv, nullptr, Vtb, nullptr, TOK, DMODEL, DMODEL);

  // Attention
  attn_wmma_kernel<<<BATCH * NHEAD * (SEQ / 16), 32, 0, stream>>>(Qb, Kb, Vtb, mask, ctxb);

  // y1 = x + ctx @ Wo + bo
  gemm_wmma_kernel<EPI_RESID><<<gProj, 256, 0, stream>>>(ctxb, Wot, bo, x, nullptr, y1, TOK, DMODEL, DMODEL);

  // LN2
  ln_bf16_kernel<<<TOK, 256, 0, stream>>>(y1, alpha2, beta2, xnb);

  // h = relu(z @ W1 + b1)
  dim3 gFF1(DFF / 64, TOK / 64);
  gemm_wmma_kernel<EPI_RELU><<<gFF1, 256, 0, stream>>>(xnb, W1t, b1, nullptr, hb, nullptr, TOK, DFF, DMODEL);

  // out = y1 + h @ W2 + b2
  gemm_wmma_kernel<EPI_RESID><<<gProj, 256, 0, stream>>>(hb, W2t, b2, y1, nullptr, out, TOK, DMODEL, DFF);
}
